// FixLocator_71528385348003
// MI455X (gfx1250) — matrix operand order repaired and checked by reading, not compiled, not run on hardware
//
#include <hip/hip_runtime.h>
#include <hip/hip_bf16.h>
#include <cstdint>
#include <cstddef>

typedef __attribute__((ext_vector_type(16))) _Float16 v16h;
typedef __attribute__((ext_vector_type(8)))  _Float16 v8h;
typedef __attribute__((ext_vector_type(8)))  float    v8f;
typedef __attribute__((ext_vector_type(4)))  int      v4i;

#define NNODE 4000
#define MPAD  4096     // NNODE rounded up to 256-row GEMM block tile
#define TG    32
#define TT    16
#define FF    256
#define HSZ   384      // H
#define H3    1152     // 3*H
#define H4    1536     // 4*H (tree: iou ++ f)
#define HS    128      // HS
#define DD4   512      // 4*HS
#define NEDGE 64000

#define BROW  40       // padded LDS row stride in f16 (80B = 20 dwords: conflict-free)

#if __has_builtin(__builtin_amdgcn_global_load_async_to_lds_b128) && \
    __has_builtin(__builtin_amdgcn_s_wait_asynccnt)
#define USE_ASYNC_LDS 1
#else
#define USE_ASYNC_LDS 0
#endif

__device__ __forceinline__ float sigm(float x) { return 1.0f / (1.0f + __expf(-x)); }

#define SHUF16(lo, hi) __builtin_shufflevector( \
    lo, hi, 0, 1, 2, 3, 4, 5, 6, 7, 8, 9, 10, 11, 12, 13, 14, 15)

#if USE_ASYNC_LDS
// Async DMA-style 16B/lane copy, global -> LDS, tracked by ASYNCcnt.
// Builtin signature (from compiler diagnostics): (AS1 v4i*, AS3 v4i*, imm, imm).
__device__ __forceinline__ void async_b128(const _Float16* g, _Float16* l)
{
    __builtin_amdgcn_global_load_async_to_lds_b128(
        (__attribute__((address_space(1))) v4i*)(uintptr_t)g,
        (__attribute__((address_space(3))) v4i*)(unsigned)(uintptr_t)l,
        0, 0);
}
#endif

// ---------------------------------------------------------------------------
// WMMA GEMM: C[M, Nn] = A[M, K] * B^T + bias,  B row-major [Nn, K] (weights)
// A, B f16; C f32 (optional f16 mirror). Nn % 64 == 0, K % 32 == 0.
// A must have >= gridDim.y*256 valid (zero-padded) rows; every lda % 8 == 0.
// Block = 256 threads (8 waves); block tile 256(M) x 64(N); wave tile 32 x 64.
// B tile double-buffered through LDS; staging uses GLOBAL_LOAD_ASYNC_TO_LDS
// when the toolchain exposes it, else an early guarded register prefetch.
// ---------------------------------------------------------------------------
__global__ __launch_bounds__(256)
void gemm_wmma(const _Float16* __restrict__ A, int lda,
               const _Float16* __restrict__ B, int ldb,
               const float* __restrict__ bias,
               float* __restrict__ C, int ldc, int ccol0,
               _Float16* __restrict__ C16, int ldc16,
               int M, int Nn, int K)
{
    __shared__ __align__(16) _Float16 Bsm[2][64 * BROW];

    const int tid   = threadIdx.x;
    const int wave  = tid >> 5;
    const int lane  = tid & 31;
    const int r     = lane & 15;
    const int half  = lane >> 4;
    const int mbase = blockIdx.y * 256 + wave * 32;
    const int nbase = blockIdx.x * 64;

    // Cooperative B staging: thread -> (col, 8-f16 segment)
    const int bcol = tid >> 2;   // 0..63
    const int bseg = tid & 3;    // 0..3
    const _Float16* Bg = B + (size_t)(nbase + bcol) * ldb + bseg * 8;
    _Float16*       Bs = &Bsm[0][0] + bcol * BROW + bseg * 8;

    v8f acc[2][4] = {};

    // A fragment = two contiguous 16-byte runs per lane (CDNA5 16-bit A layout:
    // lane-half h covers K = {8h..8h+7} and {16+8h..16+8h+7} of each 32-chunk).
    const _Float16* Ap0 = A + (size_t)(mbase + r) * lda + (half << 3);
    const _Float16* Ap1 = Ap0 + (size_t)16 * lda;

    const int nchunk = K >> 5;

    // Prologue: stage chunk 0 into buffer 0.
#if USE_ASYNC_LDS
    async_b128(Bg, Bs);
    __builtin_amdgcn_s_wait_asynccnt(0);
#else
    *(v8h*)Bs = *(const v8h*)Bg;
#endif
    __syncthreads();

    for (int i = 0; i < nchunk; ++i) {
        const int  k0   = i << 5;
        const int  cur  = i & 1;
        const bool more = (i + 1) < nchunk;

#if USE_ASYNC_LDS
        // Issue the async copy for the next chunk before computing this one;
        // completion is awaited just before the barrier.
        if (more) async_b128(Bg + k0 + 32, Bs + (1 - cur) * (64 * BROW));
#else
        v8h pre = {};
        if (more) pre = *(const v8h*)(Bg + k0 + 32);
#endif

        // A fragments for this wave's two 16-row subtiles.
        const v16h a0 = SHUF16(*(const v8h*)(Ap0 + k0), *(const v8h*)(Ap0 + k0 + 16));
        const v16h a1 = SHUF16(*(const v8h*)(Ap1 + k0), *(const v8h*)(Ap1 + k0 + 16));

        v16h bf[4];
#pragma unroll
        for (int t = 0; t < 4; ++t)
            bf[t] = SHUF16(*(const v8h*)(&Bsm[cur][(t * 16 + r) * BROW + (half << 3)]),
                           *(const v8h*)(&Bsm[cur][(t * 16 + r) * BROW + 16 + (half << 3)]));

#pragma unroll
        for (int t = 0; t < 4; ++t) {
            acc[0][t] = __builtin_amdgcn_wmma_f32_16x16x32_f16(
                false, a0, false, bf[t], (short)0, acc[0][t], false, false);
            acc[1][t] = __builtin_amdgcn_wmma_f32_16x16x32_f16(
                false, a1, false, bf[t], (short)0, acc[1][t], false, false);
        }

#if USE_ASYNC_LDS
        if (more) __builtin_amdgcn_s_wait_asynccnt(0);
#else
        if (more) *(v8h*)(Bs + (1 - cur) * (64 * BROW)) = pre;
#endif
        __syncthreads();
    }

#pragma unroll
    for (int t = 0; t < 4; ++t) {
        const int   ncol = nbase + t * 16 + r;
        const float bv   = bias ? bias[ncol] : 0.0f;
#pragma unroll
        for (int s = 0; s < 2; ++s) {
#pragma unroll
            for (int j = 0; j < 8; ++j) {
                const int row = mbase + s * 16 + j + (half << 3);
                if (row < M) {
                    const float val = acc[s][t][j] + bv;
                    C[(size_t)row * ldc + ccol0 + ncol] = val;
                    if (C16) C16[(size_t)row * ldc16 + ccol0 + ncol] = (_Float16)val;
                }
            }
        }
    }
}

// ---------------------------------------------------------------------------
// Elementwise / utility kernels
// ---------------------------------------------------------------------------
__global__ void cvt_f16(const float* __restrict__ s, _Float16* __restrict__ d, size_t n)
{
    size_t i = (size_t)blockIdx.x * 256 + threadIdx.x;
    if (i < n) d[i] = (_Float16)s[i];
}

__global__ void cat2_f16(const float* __restrict__ a, size_t na,
                         const float* __restrict__ b, size_t nb,
                         _Float16* __restrict__ d)
{
    size_t i = (size_t)blockIdx.x * 256 + threadIdx.x;
    if (i < na)           d[i] = (_Float16)a[i];
    else if (i < na + nb) d[i] = (_Float16)b[i - na];
}

__global__ void cat2_f32(const float* __restrict__ a, size_t na,
                         const float* __restrict__ b, size_t nb,
                         float* __restrict__ d)
{
    size_t i = (size_t)blockIdx.x * 256 + threadIdx.x;
    if (i < na)           d[i] = a[i];
    else if (i < na + nb) d[i] = b[i - na];
}

__global__ void fill_f32(float* __restrict__ p, float v, size_t n)
{
    size_t i = (size_t)blockIdx.x * 256 + threadIdx.x;
    if (i < n) p[i] = v;
}

__global__ void fill_f16(_Float16* __restrict__ p, size_t n)
{
    size_t i = (size_t)blockIdx.x * 256 + threadIdx.x;
    if (i < n) p[i] = (_Float16)0.0f;
}

// GRU gates: XG = x_t@Wi^T + bi, HG = h@Wh^T + bh (both [N, 3H], order r,z,n)
__global__ void gru_gate(const float* __restrict__ XG, const float* __restrict__ HG,
                         float* __restrict__ h, _Float16* __restrict__ h16)
{
    int idx = blockIdx.x * 256 + threadIdx.x;
    if (idx >= NNODE * HSZ) return;
    const int i = idx / HSZ, j = idx % HSZ;
    const float* x = XG + (size_t)i * H3;
    const float* g = HG + (size_t)i * H3;
    const float r = sigm(x[j] + g[j]);
    const float z = sigm(x[HSZ + j] + g[HSZ + j]);
    const float n = tanhf(x[2 * HSZ + j] + r * g[2 * HSZ + j]);
    const float hv = (1.0f - z) * n + z * h[idx];
    h[idx]                   = hv;
    h16[(size_t)i * HSZ + j] = (_Float16)hv;
}

// TreeLSTM gates: XG/HG [N, 4H]: cols [0,3H) = i,o,u  |  [3H,4H) = f-preact
__global__ void tree_gate(const float* __restrict__ XG, const float* __restrict__ HG,
                          float* __restrict__ c, float* __restrict__ h,
                          _Float16* __restrict__ h16)
{
    int idx = blockIdx.x * 256 + threadIdx.x;
    if (idx >= NNODE * HSZ) return;
    const int i = idx / HSZ, j = idx % HSZ;
    const float* x = XG + (size_t)i * H4;
    const float* g = HG + (size_t)i * H4;
    const float iv = x[j] + g[j];
    const float ov = x[HSZ + j] + g[HSZ + j];
    const float uv = x[2 * HSZ + j] + g[2 * HSZ + j];
    const float fv = sigm(x[3 * HSZ + j] + g[3 * HSZ + j]);
    const float cv = sigm(iv) * tanhf(uv) + fv * c[idx];
    const float hv = sigm(ov) * tanhf(cv);
    c[idx]                   = cv;
    h[idx]                   = hv;
    h16[(size_t)i * HSZ + j] = (_Float16)hv;
}

// GCN normalization helpers
__global__ void deg_count(const int* __restrict__ col, float* __restrict__ deg)
{
    int e = blockIdx.x * 256 + threadIdx.x;
    if (e < NEDGE) atomicAdd(&deg[col[e]], 1.0f);
}

__global__ void to_dinv(float* __restrict__ deg)
{
    int i = blockIdx.x * 256 + threadIdx.x;
    if (i < NNODE) deg[i] = rsqrtf(deg[i]);   // deg >= 1 (self loop)
}

__global__ void agg_init(const float* __restrict__ xw, const float* __restrict__ dinv,
                         const float* __restrict__ bias, float* __restrict__ out)
{
    int idx = blockIdx.x * 256 + threadIdx.x;
    if (idx >= NNODE * DD4) return;
    const int i = idx >> 9, f = idx & 511;
    const float di = dinv[i];
    out[idx] = bias[f] + xw[idx] * di * di;
}

__global__ void agg_edge(const int* __restrict__ row, const int* __restrict__ col,
                         const float* __restrict__ xw, const float* __restrict__ dinv,
                         float* __restrict__ out)
{
    const int e = blockIdx.x;
    const int r = row[e], c = col[e];
    const float w = dinv[r] * dinv[c];
    const float* src = xw + (size_t)r * DD4;
    float*       dst = out + (size_t)c * DD4;
    const int f = threadIdx.x;
    atomicAdd(&dst[f],       src[f]       * w);
    atomicAdd(&dst[f + 256], src[f + 256] * w);
}

__global__ void relu_cvt(float* __restrict__ x, _Float16* __restrict__ x16, size_t n)
{
    size_t i = (size_t)blockIdx.x * 256 + threadIdx.x;
    if (i >= n) return;
    const float v = fmaxf(x[i], 0.0f);
    x[i]   = v;
    x16[i] = (_Float16)v;
}

// 512 -> 2 head + softmax, output [2, N]
__global__ void head_softmax(const float* __restrict__ m, const float* __restrict__ w,
                             const float* __restrict__ b, float* __restrict__ out)
{
    int i = blockIdx.x * 256 + threadIdx.x;
    if (i >= NNODE) return;
    float a0 = b[0], a1 = b[1];
    const float* x = m + (size_t)i * DD4;
    for (int k = 0; k < DD4; ++k) {
        const float xv = x[k];
        a0 += xv * w[k];
        a1 += xv * w[DD4 + k];
    }
    const float mx = fmaxf(a0, a1);
    const float e0 = __expf(a0 - mx), e1 = __expf(a1 - mx);
    const float s = e0 + e1;
    out[i]         = e0 / s;
    out[NNODE + i] = e1 / s;
}

// ---------------------------------------------------------------------------
extern "C" void kernel_launch(void* const* d_in, const int* in_sizes, int n_in,
                              void* d_out, int out_size, void* d_ws, size_t ws_size,
                              hipStream_t stream)
{
    (void)in_sizes; (void)n_in; (void)out_size; (void)ws_size;

    const float* feat1 = (const float*)d_in[0];
    const float* feat2 = (const float*)d_in[1];
    const float* feat3 = (const float*)d_in[2];
    const float* feat4 = (const float*)d_in[3];
    const int*   eidx  = (const int*)d_in[4];
    const int*   erow  = eidx;
    const int*   ecol  = eidx + NEDGE;

    const float* gru_wi = (const float*)d_in[5];
    const float* gru_wh = (const float*)d_in[6];
    const float* gru_bi = (const float*)d_in[7];
    const float* gru_bh = (const float*)d_in[8];

    const float* t_wx[2][2] = { {(const float*)d_in[9],  (const float*)d_in[13]},
                                {(const float*)d_in[17], (const float*)d_in[21]} };
    const float* t_bx[2][2] = { {(const float*)d_in[10], (const float*)d_in[14]},
                                {(const float*)d_in[18], (const float*)d_in[22]} };
    const float* t_wh[2][2] = { {(const float*)d_in[11], (const float*)d_in[15]},
                                {(const float*)d_in[19], (const float*)d_in[23]} };
    const float* t_bh[2][2] = { {(const float*)d_in[12], (const float*)d_in[16]},
                                {(const float*)d_in[20], (const float*)d_in[24]} };

    const float* r1_w = (const float*)d_in[25]; const float* r1_b = (const float*)d_in[26];
    const float* r2_w = (const float*)d_in[27]; const float* r2_b = (const float*)d_in[28];
    const float* r3_w = (const float*)d_in[29]; const float* r3_b = (const float*)d_in[30];
    const float* r4_w = (const float*)d_in[31]; const float* r4_b = (const float*)d_in[32];
    const float* g0_w = (const float*)d_in[33]; const float* g0_b = (const float*)d_in[34];
    const float* g1_w = (const float*)d_in[35]; const float* g1_b = (const float*)d_in[36];
    const float* r7_w = (const float*)d_in[37]; const float* r7_b = (const float*)d_in[38];

    // ---- carve workspace (256-byte aligned chunks) ----
    char*  base = (char*)d_ws;
    size_t off  = 0;
    auto alloc = [&](size_t bytes) -> void* {
        off = (off + 255) & ~(size_t)255;
        void* p = base + off;
        off += bytes;
        return p;
    };

    _Float16* wi16 = (_Float16*)alloc((size_t)H3 * FF * 2);
    _Float16* wh16 = (_Float16*)alloc((size_t)H3 * HSZ * 2);
    _Float16* twx16[2]; _Float16* twh16[2]; float* tbx[2]; float* tbh[2];
    for (int s = 0; s < 2; ++s) {
        twx16[s] = (_Float16*)alloc((size_t)H4 * FF * 2);
        twh16[s] = (_Float16*)alloc((size_t)H4 * HSZ * 2);
        tbx[s]   = (float*)alloc((size_t)H4 * 4);
        tbh[s]   = (float*)alloc((size_t)H4 * 4);
    }
    _Float16* r1w16 = (_Float16*)alloc((size_t)HS * HSZ * 2);
    _Float16* r2w16 = (_Float16*)alloc((size_t)HS * HSZ * 2);
    _Float16* r3w16 = (_Float16*)alloc((size_t)HS * HSZ * 2);
    _Float16* r4w16 = (_Float16*)alloc((size_t)HS * DD4 * 2);
    _Float16* g0w16 = (_Float16*)alloc((size_t)DD4 * DD4 * 2);
    _Float16* g1w16 = (_Float16*)alloc((size_t)DD4 * DD4 * 2);

    // A-side activation buffers: padded to MPAD rows (zero tail, no guards in GEMM)
    _Float16* f1_16 = (_Float16*)alloc((size_t)MPAD * TG * FF * 2);
    _Float16* f2_16 = (_Float16*)alloc((size_t)MPAD * TT * FF * 2);
    _Float16* f3_16 = (_Float16*)alloc((size_t)MPAD * TT * FF * 2);
    _Float16* f4_16 = (_Float16*)alloc((size_t)MPAD * DD4 * 2);
    _Float16* h16   = (_Float16*)alloc((size_t)MPAD * HSZ * 2);
    _Float16* m16   = (_Float16*)alloc((size_t)MPAD * DD4 * 2);

    float* XG   = (float*)alloc((size_t)NNODE * H4 * 4);
    float* HG   = (float*)alloc((size_t)NNODE * H4 * 4);
    float* hbuf = (float*)alloc((size_t)NNODE * HSZ * 4);
    float* cbuf = (float*)alloc((size_t)NNODE * HSZ * 4);
    float* mbuf = (float*)alloc((size_t)NNODE * DD4 * 4);
    float* xw   = (float*)alloc((size_t)NNODE * DD4 * 4);
    float* buf2 = (float*)alloc((size_t)NNODE * DD4 * 4);
    float* deg  = (float*)alloc((size_t)NNODE * 4);

    auto G1 = [](size_t n) { return dim3((unsigned)((n + 255) / 256)); };
#define GEMM(A, lda, B, ldb, bias, C, ldc, ccol0, C16, ldc16, M, Nn, K)                 \
    gemm_wmma<<<dim3((unsigned)((Nn) / 64), (unsigned)(((M) + 255) / 256)), 256, 0,     \
                stream>>>(A, lda, B, ldb, bias, C, ldc, ccol0, C16, ldc16, M, Nn, K)

    // ---- zero the padded tail rows of every A-side buffer ----
    const size_t tail = (size_t)(MPAD - NNODE);
    fill_f16<<<G1(tail * TG * FF), 256, 0, stream>>>(f1_16 + (size_t)NNODE * TG * FF, tail * TG * FF);
    fill_f16<<<G1(tail * TT * FF), 256, 0, stream>>>(f2_16 + (size_t)NNODE * TT * FF, tail * TT * FF);
    fill_f16<<<G1(tail * TT * FF), 256, 0, stream>>>(f3_16 + (size_t)NNODE * TT * FF, tail * TT * FF);
    fill_f16<<<G1(tail * DD4), 256, 0, stream>>>(f4_16 + (size_t)NNODE * DD4, tail * DD4);
    fill_f16<<<G1(tail * HSZ), 256, 0, stream>>>(h16 + (size_t)NNODE * HSZ, tail * HSZ);
    fill_f16<<<G1(tail * DD4), 256, 0, stream>>>(m16 + (size_t)NNODE * DD4, tail * DD4);

    // ---- weight / activation conversion ----
    cvt_f16<<<G1((size_t)H3 * FF), 256, 0, stream>>>(gru_wi, wi16, (size_t)H3 * FF);
    cvt_f16<<<G1((size_t)H3 * HSZ), 256, 0, stream>>>(gru_wh, wh16, (size_t)H3 * HSZ);
    for (int s = 0; s < 2; ++s) {
        cat2_f16<<<G1((size_t)H4 * FF), 256, 0, stream>>>(t_wx[s][0], (size_t)H3 * FF,
                                                          t_wx[s][1], (size_t)HSZ * FF, twx16[s]);
        cat2_f16<<<G1((size_t)H4 * HSZ), 256, 0, stream>>>(t_wh[s][0], (size_t)H3 * HSZ,
                                                           t_wh[s][1], (size_t)HSZ * HSZ, twh16[s]);
        cat2_f32<<<G1(H4), 256, 0, stream>>>(t_bx[s][0], (size_t)H3, t_bx[s][1], (size_t)HSZ, tbx[s]);
        cat2_f32<<<G1(H4), 256, 0, stream>>>(t_bh[s][0], (size_t)H3, t_bh[s][1], (size_t)HSZ, tbh[s]);
    }
    cvt_f16<<<G1((size_t)HS * HSZ), 256, 0, stream>>>(r1_w, r1w16, (size_t)HS * HSZ);
    cvt_f16<<<G1((size_t)HS * HSZ), 256, 0, stream>>>(r2_w, r2w16, (size_t)HS * HSZ);
    cvt_f16<<<G1((size_t)HS * HSZ), 256, 0, stream>>>(r3_w, r3w16, (size_t)HS * HSZ);
    cvt_f16<<<G1((size_t)HS * DD4), 256, 0, stream>>>(r4_w, r4w16, (size_t)HS * DD4);
    cvt_f16<<<G1((size_t)DD4 * DD4), 256, 0, stream>>>(g0_w, g0w16, (size_t)DD4 * DD4);
    cvt_f16<<<G1((size_t)DD4 * DD4), 256, 0, stream>>>(g1_w, g1w16, (size_t)DD4 * DD4);
    cvt_f16<<<G1((size_t)NNODE * TG * FF), 256, 0, stream>>>(feat1, f1_16, (size_t)NNODE * TG * FF);
    cvt_f16<<<G1((size_t)NNODE * TT * FF), 256, 0, stream>>>(feat2, f2_16, (size_t)NNODE * TT * FF);
    cvt_f16<<<G1((size_t)NNODE * TT * FF), 256, 0, stream>>>(feat3, f3_16, (size_t)NNODE * TT * FF);
    cvt_f16<<<G1((size_t)NNODE * DD4), 256, 0, stream>>>(feat4, f4_16, (size_t)NNODE * DD4);

    const size_t nh = (size_t)NNODE * HSZ;

    // ---- GRU scan ----
    fill_f32<<<G1(nh), 256, 0, stream>>>(hbuf, 0.0f, nh);
    fill_f16<<<G1(nh), 256, 0, stream>>>(h16, nh);
    for (int t = 0; t < TG; ++t) {
        GEMM(f1_16 + (size_t)t * FF, TG * FF, wi16, FF, gru_bi,
             XG, H3, 0, (_Float16*)nullptr, 0, NNODE, H3, FF);
        GEMM(h16, HSZ, wh16, HSZ, gru_bh, HG, H3, 0, (_Float16*)nullptr, 0, NNODE, H3, HSZ);
        gru_gate<<<G1(nh), 256, 0, stream>>>(XG, HG, hbuf, h16);
    }
    GEMM(h16, HSZ, r1w16, HSZ, r1_b, mbuf, DD4, 0, (_Float16*)nullptr, 0, NNODE, HS, HSZ);

    // ---- two chain-TreeLSTM scans ----
    const _Float16* tf16[2] = { f2_16, f3_16 };
    _Float16* rw16[2] = { r2w16, r3w16 };
    const float* rb[2] = { r2_b, r3_b };
    for (int s = 0; s < 2; ++s) {
        fill_f32<<<G1(nh), 256, 0, stream>>>(hbuf, 0.0f, nh);
        fill_f32<<<G1(nh), 256, 0, stream>>>(cbuf, 0.0f, nh);
        fill_f16<<<G1(nh), 256, 0, stream>>>(h16, nh);
        for (int t = 0; t < TT; ++t) {
            GEMM(tf16[s] + (size_t)t * FF, TT * FF, twx16[s], FF, tbx[s],
                 XG, H4, 0, (_Float16*)nullptr, 0, NNODE, H4, FF);
            GEMM(h16, HSZ, twh16[s], HSZ, tbh[s], HG, H4, 0, (_Float16*)nullptr, 0, NNODE, H4, HSZ);
            tree_gate<<<G1(nh), 256, 0, stream>>>(XG, HG, cbuf, hbuf, h16);
        }
        GEMM(h16, HSZ, rw16[s], HSZ, rb[s], mbuf, DD4, 128 * (s + 1),
             (_Float16*)nullptr, 0, NNODE, HS, HSZ);
    }
    GEMM(f4_16, DD4, r4w16, DD4, r4_b, mbuf, DD4, 384, (_Float16*)nullptr, 0, NNODE, HS, DD4);

    const size_t nm = (size_t)NNODE * DD4;
    cvt_f16<<<G1(nm), 256, 0, stream>>>(mbuf, m16, nm);

    // ---- GCN normalization (deg -> dinv) ----
    fill_f32<<<G1(NNODE), 256, 0, stream>>>(deg, 1.0f, (size_t)NNODE);
    deg_count<<<G1(NEDGE), 256, 0, stream>>>(ecol, deg);
    to_dinv<<<G1(NNODE), 256, 0, stream>>>(deg);

    // ---- GCN layer 0 (+ReLU) ----
    GEMM(m16, DD4, g0w16, DD4, (const float*)nullptr, xw, DD4, 0,
         (_Float16*)nullptr, 0, NNODE, DD4, DD4);
    agg_init<<<G1(nm), 256, 0, stream>>>(xw, deg, g0_b, buf2);
    agg_edge<<<dim3(NEDGE), 256, 0, stream>>>(erow, ecol, xw, deg, buf2);
    relu_cvt<<<G1(nm), 256, 0, stream>>>(buf2, m16, nm);

    // ---- GCN layer 1 ----
    GEMM(m16, DD4, g1w16, DD4, (const float*)nullptr, xw, DD4, 0,
         (_Float16*)nullptr, 0, NNODE, DD4, DD4);
    agg_init<<<G1(nm), 256, 0, stream>>>(xw, deg, g1_b, mbuf);
    agg_edge<<<dim3(NEDGE), 256, 0, stream>>>(erow, ecol, xw, deg, mbuf);

    // ---- head + softmax ----
    head_softmax<<<G1(NNODE), 256, 0, stream>>>(mbuf, r7_w, r7_b, (float*)d_out);
#undef GEMM
}
